// MultiHeadAttention_5557687681367
// MI455X (gfx1250) — compile-verified, matrix operand order checked by reading, not run on hardware
//
#include <hip/hip_runtime.h>

// ---------------------------------------------------------------------------
// CDNA5 / gfx1250 WMMA multi-head attention (B=2, S=2048, D=1024, H=16).
//   1) 3x projection GEMM: f32 in -> bf16 WMMA -> bf16 Q/K/V (halves traffic)
//   2) flash attention: TDM (tensor_load_to_lds + s_wait_tensorcnt) stages K
//      tiles, bf16 WMMA for QK^T and P.V, online softmax in f32 VALU
//   3) output GEMM: f32 O -> bf16 WMMA -> f32 d_out
// Workspace: 3 x 8MB bf16 + 16MB f32 = 40 MB.
// ---------------------------------------------------------------------------

typedef __attribute__((ext_vector_type(16))) __bf16 v16bf;
typedef __attribute__((ext_vector_type(8)))  float  v8f;
typedef __attribute__((ext_vector_type(4)))  unsigned int u32x4;
typedef __attribute__((ext_vector_type(4)))  int  i32x4;
typedef __attribute__((ext_vector_type(8)))  int  i32x8;

// Hardware bf16 convert (v_cvt_*bf16_f32).
__device__ __forceinline__ unsigned short f2bf(float f) {
  union { __bf16 b; unsigned short u; } x;
  x.b = (__bf16)f;
  return x.u;
}

__device__ __forceinline__ unsigned long long pack4bf(float4 f) {
  return  (unsigned long long)f2bf(f.x)
       | ((unsigned long long)f2bf(f.y) << 16)
       | ((unsigned long long)f2bf(f.z) << 32)
       | ((unsigned long long)f2bf(f.w) << 48);
}

// Load one 16x32 bf16 A/B fragment (ISA VGPR layout):
// lanes 0-15 : row = lane,    K = {0..7} u {16..23}
// lanes 16-31: row = lane-16, K = {8..15} u {24..31}
__device__ __forceinline__ v16bf ld_frag(const unsigned short* p, int stride, int lane) {
  const int r  = lane & 15;
  const int kb = (lane >> 4) << 3;
  const unsigned short* base = p + r * stride + kb;
  union { v16bf v; uint4 q[2]; } u;
  u.q[0] = *(const uint4*)(base);
  u.q[1] = *(const uint4*)(base + 16);
  return u.v;
}

__device__ __forceinline__ v8f wmma_bf16(v16bf a, v16bf b, v8f c) {
  return __builtin_amdgcn_wmma_f32_16x16x32_bf16(
      false, a, false, b, (short)0, c, false, false);
}

// ---------------------------------------------------------------------------
// Tensor Data Mover: 2D tile (64 rows x 64 bf16) global -> LDS, with LDS pad
// (8 DWORDs every 32 DWORDs => row stride 80 ushorts to spread LDS banks).
// ---------------------------------------------------------------------------
__device__ __forceinline__ void tdm_load_tile_bf16(
    unsigned lds_byte_addr, const void* gptr,
    int tensor_d0, int tensor_d1, int d0_stride_elems)
{
  const unsigned long long ga = (unsigned long long)(uintptr_t)gptr;
  u32x4 g0;
  g0[0] = 1u;                                                 // count=1
  g0[1] = lds_byte_addr;                                      // lds_addr
  g0[2] = (unsigned)(ga & 0xFFFFFFFFu);                       // global_addr lo
  g0[3] = (unsigned)((ga >> 32) & 0x01FFFFFFu) | (2u << 30);  // hi | type=2
  i32x8 g1;
  g1[0] = (1 << 16)      // data_size = 2 bytes
        | (1 << 20)      // pad_enable
        | (4 << 22)      // pad_interval: every 32 DWORDs (= one 64-elem row)
        | (7 << 25);     // pad_amount: 8 DWORDs (= 16 bf16)
  g1[1] = (tensor_d0 & 0xFFFF) << 16;
  g1[2] = ((tensor_d0 >> 16) & 0xFFFF) | ((tensor_d1 & 0xFFFF) << 16);
  g1[3] = ((tensor_d1 >> 16) & 0xFFFF) | (64 << 16);          // tile_dim0 = 64
  g1[4] = 64;                                                 // tile_dim1 = 64
  g1[5] = d0_stride_elems;                                    // dim0 stride lo
  g1[6] = 0;
  g1[7] = 0;
  const i32x4 z4 = {0, 0, 0, 0};
#if __clang_major__ >= 23
  const i32x8 z8 = {0, 0, 0, 0, 0, 0, 0, 0};
  __builtin_amdgcn_tensor_load_to_lds(g0, g1, z4, z4, z8, 0);
#else
  __builtin_amdgcn_tensor_load_to_lds(g0, g1, z4, z4, 0);
#endif
}

// ---------------------------------------------------------------------------
// GEMM:  C[M,N] = A[M,K] @ W[N,K]^T + bias[N]  (f32 in, OutT out)
// 256 threads / 8 waves, 128x128 C tile, 32-wide bf16 K slabs in LDS.
// ---------------------------------------------------------------------------
#define GT_M 128
#define GT_N 128
#define GT_K 32
#define GLDK 48

template <typename OutT>
__global__ __launch_bounds__(256) void gemm_bias_wmma(
    const float* __restrict__ A, const float* __restrict__ W,
    const float* __restrict__ bias, OutT* __restrict__ C,
    int M, int N, int K)
{
  __shared__ alignas(16) unsigned short sA[GT_M][GLDK];
  __shared__ alignas(16) unsigned short sB[GT_N][GLDK];

  const int tid  = threadIdx.x;
  const int lane = tid & 31;
  const int wid  = tid >> 5;
  const int wm   = wid >> 2;
  const int wn   = wid & 3;
  const size_t blockM = (size_t)blockIdx.y * GT_M;
  const size_t blockN = (size_t)blockIdx.x * GT_N;

  v8f zero8 = {0.f,0.f,0.f,0.f,0.f,0.f,0.f,0.f};
  v8f acc[4][2];
  #pragma unroll
  for (int mt = 0; mt < 4; ++mt)
    #pragma unroll
    for (int nt = 0; nt < 2; ++nt) acc[mt][nt] = zero8;

  for (int k0 = 0; k0 < K; k0 += GT_K) {
    for (int i = tid; i < (GT_M * GT_K) / 4; i += 256) {
      const int r  = i >> 3;
      const int c4 = (i & 7) << 2;
      float4 fa = *(const float4*)(A + (blockM + r) * (size_t)K + k0 + c4);
      *(unsigned long long*)&sA[r][c4] = pack4bf(fa);
      float4 fb = *(const float4*)(W + (blockN + r) * (size_t)K + k0 + c4);
      *(unsigned long long*)&sB[r][c4] = pack4bf(fb);
    }
    __syncthreads();

    if (k0 + GT_K < K) {
      __builtin_prefetch(A + (blockM + (tid >> 1)) * (size_t)K + k0 + GT_K + ((tid & 1) << 4), 0, 1);
      __builtin_prefetch(W + (blockN + (tid >> 1)) * (size_t)K + k0 + GT_K + ((tid & 1) << 4), 0, 1);
    }

    v16bf af[4], bf2[2];
    #pragma unroll
    for (int mt = 0; mt < 4; ++mt)
      af[mt] = ld_frag(&sA[wm * 64 + mt * 16][0], GLDK, lane);
    #pragma unroll
    for (int nt = 0; nt < 2; ++nt)
      bf2[nt] = ld_frag(&sB[wn * 32 + nt * 16][0], GLDK, lane);

    #pragma unroll
    for (int mt = 0; mt < 4; ++mt)
      #pragma unroll
      for (int nt = 0; nt < 2; ++nt)
        acc[mt][nt] = wmma_bf16(af[mt], bf2[nt], acc[mt][nt]);
    __syncthreads();
  }

  #pragma unroll
  for (int mt = 0; mt < 4; ++mt) {
    #pragma unroll
    for (int nt = 0; nt < 2; ++nt) {
      const size_t row0 = blockM + wm * 64 + mt * 16 + ((lane >> 4) << 3);
      const size_t col  = blockN + wn * 32 + nt * 16 + (lane & 15);
      const float bv = bias[col];
      #pragma unroll
      for (int r = 0; r < 8; ++r)
        C[(row0 + r) * (size_t)N + col] = (OutT)(acc[mt][nt][r] + bv);
    }
  }
}

// ---------------------------------------------------------------------------
// Flash attention, causal. Q/K/V are bf16 [B, S, H*64] (head h = cols h*64..).
// WG = one (b,h) x 128 q rows; 8 waves x 16 q rows. K tile staged by TDM
// (wave 0, s_wait_tensorcnt), V tile transposed by 16-bit moves.
// All LDS fragments are preloaded into register arrays before the WMMA
// chains so the tensor pipe overlaps ds_load latency (partial dscnt waits).
// ---------------------------------------------------------------------------
#define FA_KB  64
#define FA_LDK 80   // matches TDM pad: 64 data + 16 pad ushorts per row

__global__ __launch_bounds__(256) void flash_attn_wmma(
    const unsigned short* __restrict__ Qp, const unsigned short* __restrict__ Kp,
    const unsigned short* __restrict__ Vp, float* __restrict__ Op,
    int Bn, int Hn, int Sn)
{
  constexpr int DEP = 64;
  const int Dm = Hn * DEP;

  __shared__ alignas(16) unsigned short sK [FA_KB][FA_LDK];   // [key][d], TDM dest
  __shared__ alignas(16) unsigned short sVt[DEP  ][FA_LDK];   // [d][key]
  __shared__ alignas(16) unsigned short sP [8][16][FA_LDK];   // per-wave P tile

  const int tid = threadIdx.x, lane = tid & 31, wid = tid >> 5;
  const int b  = blockIdx.z, h = blockIdx.y;
  const int q0 = blockIdx.x * 128;
  const int qw = q0 + wid * 16;

  const unsigned short* Kb = Kp + ((size_t)b * Sn) * Dm + h * DEP;
  const unsigned short* Vb = Vp + ((size_t)b * Sn) * Dm + h * DEP;
  float*                Ob = Op + ((size_t)b * Sn) * Dm + h * DEP;

  // Q fragments straight from global bf16 (global_load_b128 per half).
  const unsigned short* Qrow = Qp + ((size_t)b * Sn + qw) * Dm + h * DEP;
  v16bf qf[2];
  qf[0] = ld_frag(Qrow,      Dm, lane);
  qf[1] = ld_frag(Qrow + 32, Dm, lane);

  v8f zero8 = {0.f,0.f,0.f,0.f,0.f,0.f,0.f,0.f};
  v8f oacc[4];
  #pragma unroll
  for (int nt = 0; nt < 4; ++nt) oacc[nt] = zero8;
  float rm[8], rl[8];
  #pragma unroll
  for (int r = 0; r < 8; ++r) { rm[r] = -3.0e38f; rl[r] = 0.0f; }

  const unsigned sK_lds = (unsigned)(uintptr_t)&sK[0][0];

  for (int kb = 0; kb < q0 + 128; kb += FA_KB) {   // causal upper bound
    // K tile: one TDM descriptor per block-step (issued by wave 0).
    if (wid == 0)
      tdm_load_tile_bf16(sK_lds, Kb + (size_t)kb * Dm, DEP, Sn - kb, Dm);

    // V tile transposed (d-major) with pure 16-bit moves, all threads.
    for (int i = tid; i < (FA_KB * DEP) / 4; i += 256) {
      const int r = i >> 4;
      const int c = (i & 15) << 2;
      uint2 w = *(const uint2*)(Vb + (size_t)(kb + r) * Dm + c);
      sVt[c + 0][r] = (unsigned short)(w.x & 0xFFFFu);
      sVt[c + 1][r] = (unsigned short)(w.x >> 16);
      sVt[c + 2][r] = (unsigned short)(w.y & 0xFFFFu);
      sVt[c + 3][r] = (unsigned short)(w.y >> 16);
    }
    if (wid == 0)
      __builtin_amdgcn_s_wait_tensorcnt(0);
    __syncthreads();

    if (qw + 15 >= kb) {
      // Preload all 8 K-tile B-fragments, then run the 8-WMMA chain so the
      // scheduler can overlap ds_loads with WMMA issue.
      v16bf kf[4][2];
      #pragma unroll
      for (int nt = 0; nt < 4; ++nt)
        #pragma unroll
        for (int j = 0; j < 2; ++j)
          kf[nt][j] = ld_frag(&sK[nt * 16][0] + j * 32, FA_LDK, lane);

      v8f sacc[4];
      #pragma unroll
      for (int nt = 0; nt < 4; ++nt) sacc[nt] = zero8;
      #pragma unroll
      for (int nt = 0; nt < 4; ++nt)
        #pragma unroll
        for (int j = 0; j < 2; ++j)
          sacc[nt] = wmma_bf16(qf[j], kf[nt][j], sacc[nt]);

      // scale by 1/sqrt(64) + causal mask
      const int kcol  = kb + (lane & 15);
      const int qrow0 = qw + ((lane >> 4) << 3);
      #pragma unroll
      for (int nt = 0; nt < 4; ++nt)
        #pragma unroll
        for (int r = 0; r < 8; ++r) {
          float v = sacc[nt][r] * 0.125f;
          sacc[nt][r] = (kcol + nt * 16 > qrow0 + r) ? -1.0e30f : v;
        }

      // online softmax (16-lane shfl_xor butterflies within each half-wave)
      #pragma unroll
      for (int r = 0; r < 8; ++r) {
        float m = fmaxf(fmaxf(sacc[0][r], sacc[1][r]), fmaxf(sacc[2][r], sacc[3][r]));
        #pragma unroll
        for (int off = 1; off < 16; off <<= 1)
          m = fmaxf(m, __shfl_xor(m, off, 32));
        const float mnew = fmaxf(rm[r], m);
        const float corr = __expf(rm[r] - mnew);
        rm[r] = mnew;
        float ssum = 0.0f;
        #pragma unroll
        for (int nt = 0; nt < 4; ++nt) {
          const float e = __expf(sacc[nt][r] - mnew);
          sacc[nt][r] = e;
          ssum += e;
          oacc[nt][r] *= corr;
        }
        #pragma unroll
        for (int off = 1; off < 16; off <<= 1)
          ssum += __shfl_xor(ssum, off, 32);
        rl[r] = rl[r] * corr + ssum;
      }

      // P -> per-wave LDS bf16 tile, then O += P.V (fragments preloaded too)
      {
        const int rbase = (lane >> 4) << 3;
        const int cbase = lane & 15;
        #pragma unroll
        for (int nt = 0; nt < 4; ++nt)
          #pragma unroll
          for (int r = 0; r < 8; ++r)
            sP[wid][rbase + r][nt * 16 + cbase] = f2bf(sacc[nt][r]);
      }
      v16bf pf[2];
      pf[0] = ld_frag(&sP[wid][0][0],      FA_LDK, lane);
      pf[1] = ld_frag(&sP[wid][0][0] + 32, FA_LDK, lane);
      v16bf vf[4][2];
      #pragma unroll
      for (int nt = 0; nt < 4; ++nt)
        #pragma unroll
        for (int j = 0; j < 2; ++j)
          vf[nt][j] = ld_frag(&sVt[nt * 16][0] + j * 32, FA_LDK, lane);
      #pragma unroll
      for (int nt = 0; nt < 4; ++nt)
        #pragma unroll
        for (int j = 0; j < 2; ++j)
          oacc[nt] = wmma_bf16(pf[j], vf[nt][j], oacc[nt]);
    }
    __syncthreads();
  }

  const int rbase = (lane >> 4) << 3;
  const int cbase = lane & 15;
  #pragma unroll
  for (int nt = 0; nt < 4; ++nt)
    #pragma unroll
    for (int r = 0; r < 8; ++r)
      Ob[(size_t)(qw + rbase + r) * Dm + nt * 16 + cbase] = oacc[nt][r] * (1.0f / rl[r]);
}

// ---------------------------------------------------------------------------
extern "C" void kernel_launch(void* const* d_in, const int* in_sizes, int n_in,
                              void* d_out, int out_size, void* d_ws, size_t ws_size,
                              hipStream_t stream) {
  (void)in_sizes; (void)n_in; (void)out_size; (void)ws_size;
  const float* query = (const float*)d_in[0];
  const float* key_  = (const float*)d_in[1];
  const float* value = (const float*)d_in[2];
  /* d_in[3] = causal mask; synthesized analytically in-kernel */
  const float* wq_w = (const float*)d_in[4];
  const float* wq_b = (const float*)d_in[5];
  const float* wk_w = (const float*)d_in[6];
  const float* wk_b = (const float*)d_in[7];
  const float* wv_w = (const float*)d_in[8];
  const float* wv_b = (const float*)d_in[9];
  const float* dw   = (const float*)d_in[10];
  const float* dbv  = (const float*)d_in[11];

  constexpr int Bn = 2, Sn = 2048, Dm = 1024, Hn = 16;
  constexpr int Mrows = Bn * Sn;                 // 4096
  const size_t mat = (size_t)Mrows * Dm;         // 4M elements
  unsigned short* Qbf = (unsigned short*)d_ws;   // 8 MB
  unsigned short* Kbf = Qbf + mat;               // 8 MB
  unsigned short* Vbf = Kbf + mat;               // 8 MB
  float*          Oa  = (float*)(Vbf + mat);     // 16 MB

  dim3 blk(256);
  dim3 gg(Dm / GT_N, Mrows / GT_M);              // (8, 32)
  gemm_bias_wmma<__bf16><<<gg, blk, 0, stream>>>(query, wq_w, wq_b, (__bf16*)Qbf, Mrows, Dm, Dm);
  gemm_bias_wmma<__bf16><<<gg, blk, 0, stream>>>(key_,  wk_w, wk_b, (__bf16*)Kbf, Mrows, Dm, Dm);
  gemm_bias_wmma<__bf16><<<gg, blk, 0, stream>>>(value, wv_w, wv_b, (__bf16*)Vbf, Mrows, Dm, Dm);

  dim3 ga(Sn / 128, Hn, Bn);                     // (16, 16, 2)
  flash_attn_wmma<<<ga, blk, 0, stream>>>(Qbf, Kbf, Vbf, Oa, Bn, Hn, Sn);

  gemm_bias_wmma<float><<<gg, blk, 0, stream>>>(Oa, dw, dbv, (float*)d_out, Mrows, Dm, Dm);
}
